// BiLstmCrf_47098611368417
// MI455X (gfx1250) — matrix operand order repaired and checked by reading, not compiled, not run on hardware
//
#include <hip/hip_runtime.h>
#include <hip/hip_bf16.h>
#include <stdint.h>

// Problem dims
#define S_LEN 512
#define B_SZ  64
#define E_DIM 300
#define E_PAD 320     // pad K of embedding to multiple of 32
#define H_DIM 256
#define G_DIM 1024    // 4*H
#define T_TAGS 33
#define T_PAD  48     // pad tag dim to 3 WMMA N tiles
#define SB    (S_LEN*B_SZ)

typedef __attribute__((ext_vector_type(8)))  float  v8f;
typedef __attribute__((ext_vector_type(8)))  __bf16 v8bf;
typedef __attribute__((ext_vector_type(16))) __bf16 v16bf;
typedef __attribute__((ext_vector_type(4)))  unsigned int u32x4;
typedef __attribute__((ext_vector_type(8)))  int i32x8;
typedef __attribute__((ext_vector_type(4)))  int i32x4;

static __device__ __forceinline__ float sigmoidf_(float x){ return 1.0f/(1.0f+__expf(-x)); }

// ---------------- weight convert + zero-pad to bf16 ----------------
__global__ void convert_pad_bf16(__bf16* __restrict__ dst, const float* __restrict__ src,
                                 int Ns, int Ks, int Np, int Kp){
  int idx = blockIdx.x*blockDim.x + threadIdx.x;
  int total = Np*Kp;
  if (idx >= total) return;
  int n = idx / Kp, k = idx % Kp;
  float v = (n < Ns && k < Ks) ? src[n*Ks + k] : 0.0f;
  dst[idx] = (__bf16)v;
}

// ---------------- embedding gather -> x0 [S][B][E_PAD] bf16 ----------------
__global__ void embed_gather(__bf16* __restrict__ x0, const float* __restrict__ emb,
                             const int* __restrict__ inputs){
  int idx = blockIdx.x*blockDim.x + threadIdx.x;
  const int total = SB*E_PAD;
  if (idx >= total) return;
  int e  = idx % E_PAD;
  int sb = idx / E_PAD;
  int s = sb / B_SZ, b = sb % B_SZ;
  float v = 0.0f;
  if (e < E_DIM){
    int tok = inputs[b*S_LEN + s];
    v = emb[(size_t)tok*E_DIM + e];
  }
  x0[idx] = (__bf16)v;
}

// ---------------- generic bf16 WMMA GEMM: C[M,N] = A[M,K] @ B[N,K]^T (+bias) ----------------
template<int MT, int NT>
__global__ void gemm_bf16_wmma(const __bf16* __restrict__ A, const __bf16* __restrict__ Bw,
                               float* __restrict__ C, const float* __restrict__ bias,
                               int M, int N, int K, int biasN){
  const int lane = threadIdx.x & 31;
  const int wave = threadIdx.x >> 5;
  const int wavesPerBlock = blockDim.x >> 5;
  const int gw = blockIdx.x * wavesPerBlock + wave;
  const int mTiles = M / (16*MT);
  const int nTiles = N / (16*NT);
  if (gw >= mTiles*nTiles) return;            // wave-uniform exit; EXEC stays all-ones for WMMA
  const int wm = gw / nTiles, wn = gw % nTiles;
  const int m0 = wm*16*MT, n0 = wn*16*NT;
  const int half = lane >> 4, lr = lane & 15;

  v8f acc[MT*NT] = {};

  for (int k0 = 0; k0 < K; k0 += 32){
    v16bf af[MT], bfr[NT];
    #pragma unroll
    for (int mi = 0; mi < MT; mi++){
      const __bf16* ap = A + (size_t)(m0 + 16*mi + lr)*K + k0 + half*16;
      v8bf lo = *(const v8bf*)ap;
      v8bf hi = *(const v8bf*)(ap + 8);
      af[mi] = __builtin_shufflevector(lo, hi, 0,1,2,3,4,5,6,7,8,9,10,11,12,13,14,15);
    }
    #pragma unroll
    for (int ni = 0; ni < NT; ni++){
      bfr[ni] = *(const v16bf*)(Bw + (size_t)(n0 + 16*ni + lr)*K + k0 + half*16);
    }
    #pragma unroll
    for (int mi = 0; mi < MT; mi++)
      #pragma unroll
      for (int ni = 0; ni < NT; ni++)
        acc[mi*NT+ni] = __builtin_amdgcn_wmma_f32_16x16x32_bf16(
            false, af[mi], false, bfr[ni], (short)0, acc[mi*NT+ni], false, false);
  }

  #pragma unroll
  for (int mi = 0; mi < MT; mi++)
    #pragma unroll
    for (int ni = 0; ni < NT; ni++)
      #pragma unroll
      for (int r = 0; r < 8; r++){
        int row = m0 + 16*mi + half*8 + r;
        int col = n0 + 16*ni + lr;
        float v = acc[mi*NT+ni][r];
        if (bias) v += (col < biasN ? bias[col] : 0.0f);
        C[(size_t)row*N + col] = v;
      }
}

// ---------------- persistent recurrent LSTM (one workgroup per direction) ----------------
// Each wave owns ALL FOUR gates for the same 32 hidden columns -> cell update in registers.
// pg[tt] (256 KB) is staged into LDS by the Tensor Data Mover, overlapping the WMMA phase.
// LDS: pg buffer 256 KB + h (bf16, padded) 33 KB = 289 KB (dynamic).
#define HPAD 264
#define PG_LDS_BYTES (B_SZ*G_DIM*4)
#define LSTM_SMEM (PG_LDS_BYTES + B_SZ*HPAD*2)

__global__ void lstm_layer_kernel(const float* __restrict__ pg_f, const float* __restrict__ pg_b,
                                  const __bf16* __restrict__ whh_f, const __bf16* __restrict__ whh_b,
                                  const float* __restrict__ h0_all, const float* __restrict__ c0_all,
                                  int state_base, __bf16* __restrict__ out){
  extern __shared__ char smem[];
  float*  pgLds = (float*)smem;                   // [64][1024] staged pregates for step tt
  __bf16* hsh   = (__bf16*)(smem + PG_LDS_BYTES); // [64][HPAD]

  const int d = blockIdx.x;                       // 0 = forward, 1 = backward
  const float*  pg  = d ? pg_b  : pg_f;
  const __bf16* whh = d ? whh_b : whh_f;
  const float*  h0  = h0_all + (size_t)(state_base + d)*B_SZ*H_DIM;
  const float*  c0  = c0_all + (size_t)(state_base + d)*B_SZ*H_DIM;
  const int coloff  = d * H_DIM;

  const int tid  = threadIdx.x;                   // 1024 threads = 32 waves
  const int lane = tid & 31, wave = tid >> 5;
  const int wm = wave >> 3, wn = wave & 7;        // 4 row-groups x 8 column-groups
  const int m0 = wm*16;                           // batch rows m0..m0+15
  const int half = lane >> 4, lr = lane & 15;

  const unsigned ldsPgOff = (unsigned)(uintptr_t)(void*)pgLds;  // LDS byte offset (low 32 bits)

  // This wave's hidden-column tiles: jt = 2*wn, 2*wn+1; gate rows at gate*256 + jt*16
  float c_reg[2][8];
  #pragma unroll
  for (int u = 0; u < 2; u++)
    #pragma unroll
    for (int r = 0; r < 8; r++){
      int row  = m0 + half*8 + r;
      int jcol = (2*wn + u)*16 + lr;
      c_reg[u][r] = c0[row*H_DIM + jcol];
      hsh[row*HPAD + jcol] = (__bf16)h0[row*H_DIM + jcol];
    }
  __syncthreads();

  for (int t = 0; t < S_LEN; t++){
    const int tt = d ? (S_LEN - 1 - t) : t;
    const float* pgrow = pg + (size_t)tt*B_SZ*G_DIM;

    // ---- TDM: async stage pg[tt] (64 x 1024 fp32) into LDS; overlaps the WMMA phase ----
    if (wave == 0){
      uint64_t ga = (uint64_t)(uintptr_t)pgrow;
      u32x4 g0;
      g0[0] = 1u;                                           // count=1 valid user descriptor
      g0[1] = ldsPgOff;                                     // lds_addr (bytes)
      g0[2] = (unsigned)ga;                                 // global_addr[31:0]
      g0[3] = (unsigned)((ga >> 32) & 0x1FFFFFFu) | (2u<<30); // global_addr[56:32], type=2
      i32x8 g1;
      g1[0] = (int)(2u << 16);        // data_size = 4 bytes; workgroup_mask = 0
      g1[1] = (int)(1024u << 16);     // tensor_dim0 = 1024 (bits 79:48, low half)
      g1[2] = (int)(64u << 16);       // tensor_dim1 = 64  (bits 111:80, low half)
      g1[3] = (int)(1024u << 16);     // tile_dim0 = 1024 (bits 127:112)
      g1[4] = 64;                     // tile_dim1 = 64 (bits 143:128); tile_dim2 = 0
      g1[5] = 1024;                   // tensor_dim0_stride = 1024 (bits 207:160, low)
      g1[6] = 0;
      g1[7] = 0;
      i32x4 z4 = {0,0,0,0};
      i32x8 z8 = {0,0,0,0,0,0,0,0};
      __builtin_amdgcn_tensor_load_to_lds(g0, g1, z4, z4, z8, 0);
    }

    // Opaque 32-bit zero offset: hides t-invariance of W_hh loads (prevents hoist+spill)
    // while keeping the pointer in the global address space (global_load, not flat).
    unsigned zoff = 0;
    asm volatile("" : "+s"(zoff));
    const __bf16* whh_t = whh + zoff;

    // ---- Phase A: acc[g][u] = h_{t-1} @ W_hh^T  (WMMA, K=256, K-loop rolled) ----
    v8f acc[8] = {};
    #pragma unroll 1
    for (int k0 = 0; k0 < H_DIM; k0 += 32){
      const __bf16* ap = hsh + (m0 + lr)*HPAD + k0 + half*16;
      v8bf lo = *(const v8bf*)ap;
      v8bf hi = *(const v8bf*)(ap + 8);
      v16bf a = __builtin_shufflevector(lo, hi, 0,1,2,3,4,5,6,7,8,9,10,11,12,13,14,15);
      v16bf bfr[8];
      #pragma unroll
      for (int g = 0; g < 4; g++)
        #pragma unroll
        for (int u = 0; u < 2; u++){
          int nrow = g*256 + (2*wn + u)*16 + lr;
          bfr[g*2+u] = *(const v16bf*)(whh_t + (size_t)nrow*H_DIM + k0 + half*16);
        }
      #pragma unroll
      for (int i = 0; i < 8; i++)
        acc[i] = __builtin_amdgcn_wmma_f32_16x16x32_bf16(
            false, a, false, bfr[i], (short)0, acc[i], false, false);
    }

    if (wave == 0) __builtin_amdgcn_s_wait_tensorcnt(0);  // TDM done (issuing wave)
    __syncthreads();   // hsh reads finished + staged pg visible to all waves

    // ---- Phase B: in-register LSTM cell update from LDS-staged gates ----
    #pragma unroll
    for (int u = 0; u < 2; u++)
      #pragma unroll
      for (int r = 0; r < 8; r++){
        int row  = m0 + half*8 + r;
        int jcol = (2*wn + u)*16 + lr;
        const float* pgl = pgLds + row*G_DIM + jcol;
        float gi = acc[0*2+u][r] + pgl[0];
        float gf = acc[1*2+u][r] + pgl[256];
        float gg = acc[2*2+u][r] + pgl[512];
        float go = acc[3*2+u][r] + pgl[768];
        float c = sigmoidf_(gf)*c_reg[u][r] + sigmoidf_(gi)*tanhf(gg);
        c_reg[u][r] = c;
        float h = sigmoidf_(go)*tanhf(c);
        hsh[row*HPAD + jcol] = (__bf16)h;
        out[(size_t)tt*B_SZ*512 + row*512 + coloff + jcol] = (__bf16)h;
      }
    __syncthreads();   // h_t visible + phase B done before next step's TDM overwrite
  }
}

// ---------------- CRF numerator (gold path score), one thread per batch ----------------
__global__ void crf_numerator(const float* __restrict__ em, const int* __restrict__ labels,
                              const uint8_t* __restrict__ mask, const float* __restrict__ start_t,
                              const float* __restrict__ end_t, const float* __restrict__ trans,
                              float* __restrict__ num){
  int b = threadIdx.x;
  if (b >= B_SZ) return;
  int prev = labels[b*S_LEN + 0];
  float score = start_t[prev] + em[(size_t)b*T_PAD + prev];
  int msum = mask[b*S_LEN + 0] ? 1 : 0;
  for (int t = 1; t < S_LEN; t++){
    int cur = labels[b*S_LEN + t];
    float mt = mask[b*S_LEN + t] ? 1.0f : 0.0f;
    score += (trans[prev*T_TAGS + cur] + em[((size_t)t*B_SZ + b)*T_PAD + cur]) * mt;
    msum += mask[b*S_LEN + t] ? 1 : 0;
    prev = cur;
  }
  int send = msum - 1;
  int last = labels[b*S_LEN + send];
  num[b] = score + end_t[last];
}

// ---------------- CRF forward algorithm (log partition), one block per batch ----------------
__global__ void crf_denominator(const float* __restrict__ em, const uint8_t* __restrict__ mask,
                                const float* __restrict__ start_t, const float* __restrict__ end_t,
                                const float* __restrict__ trans, float* __restrict__ den){
  __shared__ float alpha[T_TAGS];
  __shared__ float nxt[T_TAGS];
  __shared__ float tr[T_TAGS*T_TAGS];
  const int b = blockIdx.x;
  const int tid = threadIdx.x;
  for (int i = tid; i < T_TAGS*T_TAGS; i += blockDim.x) tr[i] = trans[i];
  if (tid < T_TAGS) alpha[tid] = start_t[tid] + em[(size_t)b*T_PAD + tid];
  __syncthreads();
  for (int t = 1; t < S_LEN; t++){
    if (tid < T_TAGS){
      float m = -3.4e38f;
      for (int p = 0; p < T_TAGS; p++) m = fmaxf(m, alpha[p] + tr[p*T_TAGS + tid]);
      float s = 0.0f;
      for (int p = 0; p < T_TAGS; p++) s += __expf(alpha[p] + tr[p*T_TAGS + tid] - m);
      float v = m + __logf(s) + em[((size_t)t*B_SZ + b)*T_PAD + tid];
      nxt[tid] = mask[b*S_LEN + t] ? v : alpha[tid];
    }
    __syncthreads();
    if (tid < T_TAGS) alpha[tid] = nxt[tid];
    __syncthreads();
  }
  if (tid == 0){
    float m = -3.4e38f;
    for (int c = 0; c < T_TAGS; c++) m = fmaxf(m, alpha[c] + end_t[c]);
    float s = 0.0f;
    for (int c = 0; c < T_TAGS; c++) s += __expf(alpha[c] + end_t[c] - m);
    den[b] = m + __logf(s);
  }
}

// ---------------- final loss = -mean(num - den) ----------------
__global__ void crf_loss(const float* __restrict__ num, const float* __restrict__ den,
                         float* __restrict__ out){
  __shared__ float sh[B_SZ];
  int t = threadIdx.x;
  sh[t] = num[t] - den[t];
  __syncthreads();
  for (int s = 32; s > 0; s >>= 1){
    if (t < s) sh[t] += sh[t + s];
    __syncthreads();
  }
  if (t == 0) out[0] = -sh[0] / (float)B_SZ;
}

// =======================================================================================
extern "C" void kernel_launch(void* const* d_in, const int* in_sizes, int n_in,
                              void* d_out, int out_size, void* d_ws, size_t ws_size,
                              hipStream_t stream) {
  (void)in_sizes; (void)n_in; (void)out_size; (void)ws_size;

  const int*     inputs = (const int*)d_in[0];
  const int*     labels = (const int*)d_in[1];
  const uint8_t* mask   = (const uint8_t*)d_in[2];
  const float*   emb    = (const float*)d_in[3];
  const float *w_ih_0f=(const float*)d_in[4],  *w_hh_0f=(const float*)d_in[5],  *b_0f=(const float*)d_in[6];
  const float *w_ih_0b=(const float*)d_in[7],  *w_hh_0b=(const float*)d_in[8],  *b_0b=(const float*)d_in[9];
  const float *w_ih_1f=(const float*)d_in[10], *w_hh_1f=(const float*)d_in[11], *b_1f=(const float*)d_in[12];
  const float *w_ih_1b=(const float*)d_in[13], *w_hh_1b=(const float*)d_in[14], *b_1b=(const float*)d_in[15];
  const float *lin_w=(const float*)d_in[16],   *lin_b=(const float*)d_in[17];
  const float *start_t=(const float*)d_in[18], *end_t=(const float*)d_in[19],  *trans=(const float*)d_in[20];
  const float *h0=(const float*)d_in[21],      *c0=(const float*)d_in[22];

  char* ws = (char*)d_ws;
  size_t off = 0;
  auto alloc = [&](size_t bytes) -> char* {
    char* p = ws + off;
    off += (bytes + 255) & ~(size_t)255;
    return p;
  };

  __bf16* x0     = (__bf16*)alloc((size_t)SB*E_PAD*2);        // [S][B][320]
  float*  pg0f   = (float*) alloc((size_t)SB*G_DIM*4);        // [S][B][1024]
  float*  pg0b   = (float*) alloc((size_t)SB*G_DIM*4);
  __bf16* x1     = (__bf16*)alloc((size_t)SB*512*2);          // [S][B][512] layer1 input
  float*  pg1f   = (float*) alloc((size_t)SB*G_DIM*4);
  float*  pg1b   = (float*) alloc((size_t)SB*G_DIM*4);
  __bf16* out2   = (__bf16*)alloc((size_t)SB*512*2);          // [S][B][512] layer2 output
  float*  em     = (float*) alloc((size_t)SB*T_PAD*4);        // [S][B][48] emissions (+lin_b)
  __bf16* wih0f  = (__bf16*)alloc((size_t)G_DIM*E_PAD*2);
  __bf16* wih0b  = (__bf16*)alloc((size_t)G_DIM*E_PAD*2);
  __bf16* whh0f  = (__bf16*)alloc((size_t)G_DIM*H_DIM*2);
  __bf16* whh0b  = (__bf16*)alloc((size_t)G_DIM*H_DIM*2);
  __bf16* wih1f  = (__bf16*)alloc((size_t)G_DIM*512*2);
  __bf16* wih1b  = (__bf16*)alloc((size_t)G_DIM*512*2);
  __bf16* whh1f  = (__bf16*)alloc((size_t)G_DIM*H_DIM*2);
  __bf16* whh1b  = (__bf16*)alloc((size_t)G_DIM*H_DIM*2);
  __bf16* linwp  = (__bf16*)alloc((size_t)T_PAD*512*2);
  float*  numv   = (float*) alloc(B_SZ*4);
  float*  denv   = (float*) alloc(B_SZ*4);

  (void)hipFuncSetAttribute((const void*)lstm_layer_kernel,
                            hipFuncAttributeMaxDynamicSharedMemorySize, LSTM_SMEM);

  auto conv = [&](__bf16* dst, const float* src, int Ns, int Ks, int Np, int Kp){
    int total = Np*Kp;
    convert_pad_bf16<<<(total + 255)/256, 256, 0, stream>>>(dst, src, Ns, Ks, Np, Kp);
  };
  conv(wih0f, w_ih_0f, G_DIM, E_DIM, G_DIM, E_PAD);
  conv(wih0b, w_ih_0b, G_DIM, E_DIM, G_DIM, E_PAD);
  conv(whh0f, w_hh_0f, G_DIM, H_DIM, G_DIM, H_DIM);
  conv(whh0b, w_hh_0b, G_DIM, H_DIM, G_DIM, H_DIM);
  conv(wih1f, w_ih_1f, G_DIM, 512,   G_DIM, 512);
  conv(wih1b, w_ih_1b, G_DIM, 512,   G_DIM, 512);
  conv(whh1f, w_hh_1f, G_DIM, H_DIM, G_DIM, H_DIM);
  conv(whh1b, w_hh_1b, G_DIM, H_DIM, G_DIM, H_DIM);
  conv(linwp, lin_w,   T_TAGS, 512,  T_PAD, 512);

  embed_gather<<<(SB*E_PAD + 255)/256, 256, 0, stream>>>(x0, emb, inputs);

  // Layer 0 pregates: [32768,320] @ [1024,320]^T + bias  (wave tile 64x32)
  {
    int tiles = (SB/64) * (G_DIM/32);
    gemm_bf16_wmma<4,2><<<tiles/8, 256, 0, stream>>>(x0, wih0f, pg0f, b_0f, SB, G_DIM, E_PAD, G_DIM);
    gemm_bf16_wmma<4,2><<<tiles/8, 256, 0, stream>>>(x0, wih0b, pg0b, b_0b, SB, G_DIM, E_PAD, G_DIM);
  }
  // Layer 0 recurrence (fwd + bwd in parallel) -> x1 (fwd cols 0..255, bwd cols 256..511)
  lstm_layer_kernel<<<2, 1024, LSTM_SMEM, stream>>>(pg0f, pg0b, whh0f, whh0b, h0, c0, 0, x1);

  // Layer 1 pregates: [32768,512] @ [1024,512]^T + bias
  {
    int tiles = (SB/64) * (G_DIM/32);
    gemm_bf16_wmma<4,2><<<tiles/8, 256, 0, stream>>>(x1, wih1f, pg1f, b_1f, SB, G_DIM, 512, G_DIM);
    gemm_bf16_wmma<4,2><<<tiles/8, 256, 0, stream>>>(x1, wih1b, pg1b, b_1b, SB, G_DIM, 512, G_DIM);
  }
  // Layer 1 recurrence -> out2
  lstm_layer_kernel<<<2, 1024, LSTM_SMEM, stream>>>(pg1f, pg1b, whh1f, whh1b, h0, c0, 2, out2);

  // Emissions: [32768,512] @ [48,512]^T + lin_b (only first 33 cols valid downstream)
  {
    int tiles = (SB/32) * (T_PAD/48);
    gemm_bf16_wmma<2,3><<<tiles/8, 256, 0, stream>>>(out2, linwp, em, lin_b, SB, T_PAD, 512, T_TAGS);
  }

  // CRF
  crf_numerator<<<1, 64, 0, stream>>>(em, labels, mask, start_t, end_t, trans, numv);
  crf_denominator<<<B_SZ, 64, 0, stream>>>(em, mask, start_t, end_t, trans, denv);
  crf_loss<<<1, B_SZ, 0, stream>>>(numv, denv, (float*)d_out);
}
